// MILVAD_Fusion_49967649521962
// MI455X (gfx1250) — matrix operand order repaired and checked by reading, not compiled, not run on hardware
//
#include <hip/hip_runtime.h>
#include <math.h>

// ---------------------------------------------------------------------------
// MILVAD fusion for MI455X (gfx1250, wave32, WMMA).
// Shapes: B=16 S=1024 VD=1024 TD=512 CAM=512 E=512 FD=1536 H=4 HD=128
// k = int(0.1*S) = 102 (padded to 112 for 16-row WMMA tiles)
// ---------------------------------------------------------------------------

typedef __attribute__((ext_vector_type(16))) __bf16 v16bf;
typedef __attribute__((ext_vector_type(8)))  float  v8f;

#define Bn    16
#define Sn    1024
#define VDn   1024
#define TDn   512
#define CAMn  512
#define En    512
#define FDn   1536
#define Hn    4
#define HDn   128
#define KSEL  102
#define KPAD  112

// ---------------------------------------------------------------------------
// Generic tiled bf16 WMMA GEMM: out[M,N] = act(A[M,K](lda) @ W[K,N](ldw) + b)
// Block tile 128 x (NSUB*16), K-step 32, 256 threads = 8 waves.
// Wave w owns rows w*16..w*16+15 of the tile and all NSUB column sub-tiles.
// Requirements: K % 32 == 0, N % 16 == 0, grid.x covers N exactly with
// NSUB*16-wide tiles (so no N guards anywhere -> branchless WMMA loop).
// ---------------------------------------------------------------------------
template<int NSUB>
__global__ __launch_bounds__(256)
void gemm_wmma_bf16(const float* __restrict__ A, int lda,
                    const float* __restrict__ W, int ldw,
                    const float* __restrict__ bias,
                    float* __restrict__ out, int ldo,
                    int M, int K, int N, int do_relu)
{
    constexpr int NC = NSUB * 16;            // tile width in columns
    __shared__ __bf16 a_lds[128 * 32];       // [m][k]
    __shared__ __bf16 bT_lds[NC * 32];       // [n][k]
    const int tid  = threadIdx.x;
    const int lane = tid & 31;
    const int wave = tid >> 5;
    const int m0 = blockIdx.y * 128;
    const int n0 = blockIdx.x * NC;

    v8f acc[NSUB];
    #pragma unroll
    for (int c = 0; c < NSUB; ++c) acc[c] = (v8f){0,0,0,0,0,0,0,0};

    const int kTiles = K >> 5;
    for (int kt = 0; kt < kTiles; ++kt) {
        const int k0 = kt << 5;
        // ---- stage A tile (128x32) as bf16 ----
        {
            int r  = tid >> 1;
            int cc = (tid & 1) << 4;
            int gm = m0 + r;
            __bf16* dst = &a_lds[r * 32 + cc];
            if (gm < M) {
                const float* src = A + (size_t)gm * lda + k0 + cc;
                if (kt + 1 < kTiles) __builtin_prefetch(src + 32, 0, 0);
                #pragma unroll
                for (int j = 0; j < 16; ++j) dst[j] = (__bf16)src[j];
            } else {
                #pragma unroll
                for (int j = 0; j < 16; ++j) dst[j] = (__bf16)0.0f;
            }
        }
        // ---- stage W tile (32xNC) transposed into [n][k] ----
        {
            #pragma unroll
            for (int e = tid; e < 32 * NC; e += 256) {
                int kk = e / NC;
                int nl = e % NC;
                const float* src = W + (size_t)(k0 + kk) * ldw + n0 + nl;
                if (kt + 1 < kTiles && nl == 0) __builtin_prefetch(src + 32 * ldw, 0, 0);
                bT_lds[nl * 32 + kk] = (__bf16)src[0];
            }
        }
        __syncthreads();
        // ---- A fragment (ISA 16-bit A 16x32 layout) ----
        union { v16bf v; unsigned int u[8]; } af;
        {
            int row  = wave * 16 + (lane & 15);
            int koff = (lane >= 16) ? 8 : 0;
            const unsigned short* base = (const unsigned short*)&a_lds[row * 32];
            #pragma unroll
            for (int j = 0; j < 8; ++j) {
                int kd = 2 * j + ((j >= 4) ? 8 : 0) + koff;
                af.u[j] = *(const unsigned int*)(base + kd);
            }
        }
        // ---- branchless column sub-tile WMMAs ----
        #pragma unroll
        for (int c = 0; c < NSUB; ++c) {
            union { v16bf v; unsigned int u[8]; } bf;
            int nrow = c * 16 + (lane & 15);
            int kb   = (lane >= 16) ? 8 : 0;
            const unsigned short* base = (const unsigned short*)&bT_lds[nrow * 32 + kb];
            #pragma unroll
            for (int j = 0; j < 8; ++j)
                bf.u[j] = *(const unsigned int*)(base + 2 * j + ((j >= 4) ? 8 : 0));
            acc[c] = __builtin_amdgcn_wmma_f32_16x16x32_bf16(
                false, af.v, false, bf.v, (short)0, acc[c], false, false);
        }
        __syncthreads();
    }
    // ---- store (C/D layout: vgpr r -> M = r + (lane>=16)*8, N = lane&15) ----
    #pragma unroll
    for (int c = 0; c < NSUB; ++c) {
        int gn = n0 + c * 16 + (lane & 15);
        float bv = bias ? bias[gn] : 0.0f;
        #pragma unroll
        for (int r = 0; r < 8; ++r) {
            int gm = m0 + wave * 16 + r + ((lane >= 16) ? 8 : 0);
            if (gm < M) {
                float v = acc[c][r] + bv;
                if (do_relu) v = fmaxf(v, 0.0f);
                out[(size_t)gm * ldo + gn] = v;
            }
        }
    }
}

// ---------------------------------------------------------------------------
// scores[r] = h2[r,:32] . w3 + b3
// ---------------------------------------------------------------------------
__global__ __launch_bounds__(256)
void score3_kernel(const float* __restrict__ h2, const float* __restrict__ w3,
                   const float* __restrict__ b3, float* __restrict__ scores, int M)
{
    int r = blockIdx.x * 256 + threadIdx.x;
    if (r < M) {
        const float* row = h2 + (size_t)r * 32;
        float s = 0.0f;
        #pragma unroll
        for (int j = 0; j < 32; ++j) s += row[j] * w3[j];
        scores[r] = s + b3[0];
    }
}

// ---------------------------------------------------------------------------
// top-k by exact ranking (downstream use is softmax-pooled => the set, not
// the order, matters). One block per batch, S threads.
// ---------------------------------------------------------------------------
__global__ __launch_bounds__(1024)
void topk_kernel(const float* __restrict__ scores, int* __restrict__ idx_out)
{
    __shared__ float s_lds[Sn];
    int b = blockIdx.x, t = threadIdx.x;
    s_lds[t] = scores[(size_t)b * Sn + t];
    if (t >= KSEL && t < KPAD) idx_out[b * KPAD + t] = -1;
    __syncthreads();
    float my = s_lds[t];
    int rank = 0;
    for (int j = 0; j < Sn; ++j) {
        float o = s_lds[j];
        rank += (o > my) || (o == my && j < t);
    }
    if (rank < KSEL) idx_out[b * KPAD + rank] = t;
}

// gather top-k visual rows (pad rows -> zero)
__global__ __launch_bounds__(256)
void gather_kernel(const float* __restrict__ vf, const int* __restrict__ idx,
                   float* __restrict__ topk_v)
{
    int b = blockIdx.y, i = blockIdx.x;
    int id = idx[b * KPAD + i];
    float* dst = topk_v + ((size_t)b * KPAD + i) * VDn;
    if (id >= 0) {
        const float* src = vf + ((size_t)b * Sn + id) * VDn;
        for (int j = threadIdx.x; j < VDn; j += 256) dst[j] = src[j];
    } else {
        for (int j = threadIdx.x; j < VDn; j += 256) dst[j] = 0.0f;
    }
}

// ---------------------------------------------------------------------------
// CAM aggregation: softmax over the 102 real rows, weighted sum of c.
// ---------------------------------------------------------------------------
__global__ __launch_bounds__(256)
void cam_agg_kernel(const float* __restrict__ cbuf, const float* __restrict__ agg_w,
                    const float* __restrict__ agg_b, float* __restrict__ cam)
{
    __shared__ float lg[KSEL];
    __shared__ float inv_sum;
    int b = blockIdx.x, tid = threadIdx.x;
    if (tid < KSEL) {
        const float* row = cbuf + ((size_t)b * KPAD + tid) * CAMn;
        float s = 0.0f;
        for (int j = 0; j < CAMn; ++j) s += row[j] * agg_w[j];
        lg[tid] = s + agg_b[0];
    }
    __syncthreads();
    if (tid == 0) {
        float mx = -3.4e38f;
        for (int i = 0; i < KSEL; ++i) mx = fmaxf(mx, lg[i]);
        float sm = 0.0f;
        for (int i = 0; i < KSEL; ++i) { float e = expf(lg[i] - mx); lg[i] = e; sm += e; }
        inv_sum = 1.0f / sm;
    }
    __syncthreads();
    float inv = inv_sum;
    for (int j = tid; j < CAMn; j += 256) {
        float s = 0.0f;
        for (int i = 0; i < KSEL; ++i)
            s += lg[i] * cbuf[((size_t)b * KPAD + i) * CAMn + j];
        cam[b * CAMn + j] = s * inv;
    }
}

// ---------------------------------------------------------------------------
// w1[b,h,j] = sum_d in_w[j, E + h*HD + d] * qq[b, h*HD + d]
// ---------------------------------------------------------------------------
__global__ __launch_bounds__(512)
void qhead_w1_kernel(const float* __restrict__ in_w, const float* __restrict__ qq,
                     float* __restrict__ w1)
{
    int b = blockIdx.x >> 2, h = blockIdx.x & 3;
    int j = threadIdx.x;
    const float* qh = qq + b * En + h * HDn;
    const float* wr = in_w + (size_t)j * (3 * En) + En + h * HDn;
    float s = 0.0f;
    #pragma unroll 4
    for (int d = 0; d < HDn; ++d) s += wr[d] * qh[d];
    w1[((size_t)b * Hn + h) * En + j] = s;
}

// ---------------------------------------------------------------------------
// u2pad[b, i, h] = k_w[i,:] . w1[b,h,:]   (cols 4..15 zeroed)
// lbias[b,h]    = k_b . w1[b,h,:]  +  in_b_k[h*HD:] . qq_head
// ---------------------------------------------------------------------------
__global__ __launch_bounds__(256)
void ukernel(const float* __restrict__ k_w, const float* __restrict__ k_b,
             const float* __restrict__ in_b, const float* __restrict__ qq,
             const float* __restrict__ w1, float* __restrict__ u2pad,
             float* __restrict__ lbias)
{
    int b = blockIdx.x >> 2, h = blockIdx.x & 3;
    const float* w1v = w1 + ((size_t)b * Hn + h) * En;
    for (int i = threadIdx.x; i < FDn; i += 256) {
        const float* kr = k_w + (size_t)i * En;
        float s = 0.0f;
        for (int j = 0; j < En; ++j) s += kr[j] * w1v[j];
        u2pad[((size_t)b * FDn + i) * 16 + h] = s;
        if (h == 0) {
            #pragma unroll
            for (int c = Hn; c < 16; ++c) u2pad[((size_t)b * FDn + i) * 16 + c] = 0.0f;
        }
    }
    if (threadIdx.x == 0) {
        float s = 0.0f;
        for (int j = 0; j < En; ++j) s += k_b[j] * w1v[j];
        const float* qh = qq + b * En + h * HDn;
        for (int d = 0; d < HDn; ++d) s += in_b[En + h * HDn + d] * qh[d];
        lbias[b * Hn + h] = s;
    }
}

// ---------------------------------------------------------------------------
// logits[b,h,s] = (fused[b,s,:] . u2pad[b,:,h] + lbias[b,h]) / sqrt(HD)
// Batched WMMA: per block a 128-row (s) tile, N=16 (4 heads padded), K=1536.
// vf/tf choice is uniform per 32-wide K-tile (VD % 32 == 0).
// ---------------------------------------------------------------------------
__global__ __launch_bounds__(256)
void logits_wmma_kernel(const float* __restrict__ vf, const float* __restrict__ tf,
                        const float* __restrict__ u2pad, const float* __restrict__ lbias,
                        float* __restrict__ logits)
{
    __shared__ __bf16 a_lds[128 * 32];
    __shared__ __bf16 bT_lds[16 * 32];
    const int b = blockIdx.y;
    const int m0 = blockIdx.x * 128;
    const int tid = threadIdx.x, lane = tid & 31, wave = tid >> 5;
    v8f acc = (v8f){0,0,0,0,0,0,0,0};

    for (int kt = 0; kt < FDn / 32; ++kt) {
        const int k0 = kt << 5;
        {   // stage fused tile (128 s-rows x 32 k); source uniform per tile
            int r = tid >> 1;
            int cc = (tid & 1) << 4;
            int s = m0 + r;
            const float* src = (k0 < VDn)
                ? (vf + ((size_t)b * Sn + s) * VDn + k0 + cc)
                : (tf + ((size_t)b * Sn + s) * TDn + (k0 - VDn) + cc);
            if (kt + 1 < FDn / 32) __builtin_prefetch(src + 32, 0, 0);
            __bf16* dst = &a_lds[r * 32 + cc];
            #pragma unroll
            for (int j = 0; j < 16; ++j) dst[j] = (__bf16)src[j];
        }
        {   // stage u2pad tile (32 k x 16 n) transposed
            #pragma unroll
            for (int q = 0; q < 2; ++q) {
                int e = tid * 2 + q;              // 0..511
                int kk = e >> 4, n = e & 15;
                bT_lds[n * 32 + kk] =
                    (__bf16)u2pad[((size_t)b * FDn + k0 + kk) * 16 + n];
            }
        }
        __syncthreads();
        union { v16bf v; unsigned int u[8]; } af, bf;
        {
            int row = wave * 16 + (lane & 15);
            int koff = (lane >= 16) ? 8 : 0;
            const unsigned short* base = (const unsigned short*)&a_lds[row * 32];
            #pragma unroll
            for (int j = 0; j < 8; ++j)
                af.u[j] = *(const unsigned int*)(base + 2 * j + ((j >= 4) ? 8 : 0) + koff);
        }
        {
            int nrow = lane & 15;
            int kb = (lane >= 16) ? 8 : 0;
            const unsigned short* base = (const unsigned short*)&bT_lds[nrow * 32 + kb];
            #pragma unroll
            for (int j = 0; j < 8; ++j)
                bf.u[j] = *(const unsigned int*)(base + 2 * j + ((j >= 4) ? 8 : 0));
        }
        acc = __builtin_amdgcn_wmma_f32_16x16x32_bf16(
            false, af.v, false, bf.v, (short)0, acc, false, false);
        __syncthreads();
    }
    const float inv_sqrt_hd = 0.08838834764831845f; // 1/sqrt(128)
    int h = lane & 15;
    if (h < Hn) {
        float lb = lbias[b * Hn + h];
        #pragma unroll
        for (int r = 0; r < 8; ++r) {
            int s = m0 + wave * 16 + r + ((lane >= 16) ? 8 : 0);
            logits[((size_t)b * Hn + h) * Sn + s] = (acc[r] + lb) * inv_sqrt_hd;
        }
    }
}

// ---------------------------------------------------------------------------
// softmax over S per (b,h); writes probspad[b, row(16), s]; pad rows zeroed.
// ---------------------------------------------------------------------------
__global__ __launch_bounds__(256)
void softmax_pad_kernel(const float* __restrict__ logits, float* __restrict__ probspad)
{
    int b = blockIdx.x >> 4, r = blockIdx.x & 15;
    int tid = threadIdx.x;
    float* dst = probspad + ((size_t)b * 16 + r) * Sn;
    if (r >= Hn) {
        for (int i = tid; i < Sn; i += 256) dst[i] = 0.0f;
        return;
    }
    const float* src = logits + ((size_t)b * Hn + r) * Sn;
    __shared__ float red[256];
    float mx = -3.4e38f;
    for (int i = tid; i < Sn; i += 256) mx = fmaxf(mx, src[i]);
    red[tid] = mx; __syncthreads();
    for (int o = 128; o > 0; o >>= 1) {
        if (tid < o) red[tid] = fmaxf(red[tid], red[tid + o]);
        __syncthreads();
    }
    mx = red[0]; __syncthreads();
    float sm = 0.0f;
    for (int i = tid; i < Sn; i += 256) sm += expf(src[i] - mx);
    red[tid] = sm; __syncthreads();
    for (int o = 128; o > 0; o >>= 1) {
        if (tid < o) red[tid] += red[tid + o];
        __syncthreads();
    }
    float inv = 1.0f / red[0];
    for (int i = tid; i < Sn; i += 256) dst[i] = expf(src[i] - mx) * inv;
}

// ---------------------------------------------------------------------------
// pooledF[b, r, i] = sum_s probspad[b,r,s] * fused[b,s,i]
// Batched WMMA: M=16 (heads padded), K=S=1024, N tile 128 (wave = col subtile).
// vf/tf choice uniform per block (n0 multiples of 128, VD % 128 == 0).
// ---------------------------------------------------------------------------
__global__ __launch_bounds__(256)
void pooled_wmma_kernel(const float* __restrict__ probspad,
                        const float* __restrict__ vf, const float* __restrict__ tf,
                        float* __restrict__ pooledF)
{
    __shared__ __bf16 a_lds[16 * 32];
    __shared__ __bf16 bT_lds[128 * 32];
    const int b = blockIdx.y;
    const int n0 = blockIdx.x * 128;
    const int tid = threadIdx.x, lane = tid & 31, wave = tid >> 5;
    v8f acc = (v8f){0,0,0,0,0,0,0,0};

    // uniform feature source for this block's 128 columns
    const float* fsrc = (n0 < VDn) ? vf : tf;
    const int fld     = (n0 < VDn) ? VDn : TDn;
    const int fn0     = (n0 < VDn) ? n0 : (n0 - VDn);

    for (int kt = 0; kt < Sn / 32; ++kt) {
        const int k0 = kt << 5;
        {   // stage probs tile 16x32 (2 contiguous elems/thread)
            int e = tid * 2;
            int r = e >> 5, cc = e & 31;
            const float* src = probspad + ((size_t)b * 16 + r) * Sn + k0 + cc;
            a_lds[e]     = (__bf16)src[0];
            a_lds[e + 1] = (__bf16)src[1];
        }
        {   // stage fused tile 32(s) x 128(i) transposed
            int kk = tid >> 3;
            int nl = (tid & 7) << 4;
            int s = k0 + kk;
            const float* src = fsrc + ((size_t)b * Sn + s) * fld + fn0 + nl;
            if (kt + 1 < Sn / 32) __builtin_prefetch(src + (size_t)32 * fld, 0, 0);
            #pragma unroll
            for (int j = 0; j < 16; ++j)
                bT_lds[(nl + j) * 32 + kk] = (__bf16)src[j];
        }
        __syncthreads();
        union { v16bf v; unsigned int u[8]; } af, bf;
        {
            int row = lane & 15;
            int koff = (lane >= 16) ? 8 : 0;
            const unsigned short* base = (const unsigned short*)&a_lds[row * 32];
            #pragma unroll
            for (int j = 0; j < 8; ++j)
                af.u[j] = *(const unsigned int*)(base + 2 * j + ((j >= 4) ? 8 : 0) + koff);
        }
        {
            int nrow = wave * 16 + (lane & 15);
            int kb = (lane >= 16) ? 8 : 0;
            const unsigned short* base = (const unsigned short*)&bT_lds[nrow * 32 + kb];
            #pragma unroll
            for (int j = 0; j < 8; ++j)
                bf.u[j] = *(const unsigned int*)(base + 2 * j + ((j >= 4) ? 8 : 0));
        }
        acc = __builtin_amdgcn_wmma_f32_16x16x32_bf16(
            false, af.v, false, bf.v, (short)0, acc, false, false);
        __syncthreads();
    }
    #pragma unroll
    for (int r = 0; r < 8; ++r) {
        int row = r + ((lane >= 16) ? 8 : 0);
        int gn  = n0 + wave * 16 + (lane & 15);
        pooledF[((size_t)b * 16 + row) * FDn + gn] = acc[r];
    }
}

// ---------------------------------------------------------------------------
// ctx[b, h*HD+d] = pooledVal[b*16+h, :] . in_w[:, 2E + h*HD + d] + in_b[2E+...]
// ---------------------------------------------------------------------------
__global__ __launch_bounds__(128)
void ctx_kernel(const float* __restrict__ pooledVal, const float* __restrict__ in_w,
                const float* __restrict__ in_b, float* __restrict__ ctx)
{
    int b = blockIdx.x >> 2, h = blockIdx.x & 3;
    int d = threadIdx.x;
    const float* pv = pooledVal + ((size_t)b * 16 + h) * En;
    int col = 2 * En + h * HDn + d;
    float s = 0.0f;
    for (int j = 0; j < En; ++j) s += pv[j] * in_w[(size_t)j * (3 * En) + col];
    ctx[b * En + h * HDn + d] = s + in_b[col];
}

// ---------------------------------------------------------------------------
// post-MLP once per batch (output identical at every position)
// ---------------------------------------------------------------------------
__global__ __launch_bounds__(256)
void postmlp_kernel(const float* __restrict__ attn,
                    const float* __restrict__ w1, const float* __restrict__ b1,
                    const float* __restrict__ w2, const float* __restrict__ b2,
                    const float* __restrict__ w3, const float* __restrict__ b3,
                    float* __restrict__ outval)
{
    __shared__ float g1[256];
    __shared__ float g2[32];
    int b = blockIdx.x, tid = threadIdx.x;
    const float* a = attn + b * En;
    {
        float s = 0.0f;
        for (int j = 0; j < En; ++j) s += a[j] * w1[(size_t)j * 256 + tid];
        g1[tid] = fmaxf(s + b1[tid], 0.0f);
    }
    __syncthreads();
    if (tid < 32) {
        float s = 0.0f;
        for (int j = 0; j < 256; ++j) s += g1[j] * w2[j * 32 + tid];
        g2[tid] = fmaxf(s + b2[tid], 0.0f);
    }
    __syncthreads();
    if (tid == 0) {
        float s = 0.0f;
        #pragma unroll
        for (int j = 0; j < 32; ++j) s += g2[j] * w3[j];
        outval[b] = s + b3[0];
    }
}

__global__ __launch_bounds__(256)
void bcast_kernel(const float* __restrict__ outval, float* __restrict__ out, int total)
{
    int i = blockIdx.x * 256 + threadIdx.x;
    if (i < total) out[i] = outval[i >> 10];   // S = 1024
}

// ---------------------------------------------------------------------------
extern "C" void kernel_launch(void* const* d_in, const int* in_sizes, int n_in,
                              void* d_out, int out_size, void* d_ws, size_t ws_size,
                              hipStream_t stream)
{
    const float* vf      = (const float*)d_in[0];
    const float* tf      = (const float*)d_in[1];
    const float* pre_w1  = (const float*)d_in[2];
    const float* pre_b1  = (const float*)d_in[3];
    const float* pre_w2  = (const float*)d_in[4];
    const float* pre_b2  = (const float*)d_in[5];
    const float* pre_w3  = (const float*)d_in[6];
    const float* pre_b3  = (const float*)d_in[7];
    const float* ft_w    = (const float*)d_in[8];
    const float* ft_b    = (const float*)d_in[9];
    const float* fp_w    = (const float*)d_in[10];
    const float* fp_b    = (const float*)d_in[11];
    const float* agg_w   = (const float*)d_in[12];
    const float* agg_b   = (const float*)d_in[13];
    const float* q_w     = (const float*)d_in[14];
    const float* q_b     = (const float*)d_in[15];
    const float* k_w     = (const float*)d_in[16];
    const float* k_b     = (const float*)d_in[17];
    const float* v_w     = (const float*)d_in[18];
    const float* v_b     = (const float*)d_in[19];
    const float* in_w    = (const float*)d_in[20];
    const float* in_b    = (const float*)d_in[21];
    const float* out_w   = (const float*)d_in[22];
    const float* out_b   = (const float*)d_in[23];
    const float* post_w1 = (const float*)d_in[24];
    const float* post_b1 = (const float*)d_in[25];
    const float* post_w2 = (const float*)d_in[26];
    const float* post_b2 = (const float*)d_in[27];
    const float* post_w3 = (const float*)d_in[28];
    const float* post_b3 = (const float*)d_in[29];
    (void)in_sizes; (void)n_in; (void)out_size;

    const int MROWS = Bn * Sn;      // 16384
    const int MCAM  = Bn * KPAD;    // 1792

    char* ws = (char*)d_ws;
    size_t off = 0;
    auto alloc = [&](size_t bytes) -> char* {
        char* p = ws + off;
        off += (bytes + 255) & ~(size_t)255;
        return p;
    };
    float* h1        = (float*)alloc((size_t)MROWS * 256 * 4);   // 16 MB
    float* h2        = (float*)alloc((size_t)MROWS * 32 * 4);    // 2 MB
    float* scores    = (float*)alloc((size_t)MROWS * 4);
    int*   topk_idx  = (int*)  alloc((size_t)Bn * KPAD * 4);
    float* topk_v    = (float*)alloc((size_t)MCAM * VDn * 4);    // 7 MB
    float* t_buf     = (float*)alloc((size_t)MCAM * CAMn * 4);   // 3.5 MB
    float* c_buf     = (float*)alloc((size_t)MCAM * CAMn * 4);   // 3.5 MB
    float* cam       = (float*)alloc((size_t)Bn * CAMn * 4);
    float* q1        = (float*)alloc((size_t)Bn * En * 4);
    float* qq        = (float*)alloc((size_t)Bn * En * 4);
    float* w1v       = (float*)alloc((size_t)Bn * Hn * En * 4);
    float* u2pad     = (float*)alloc((size_t)Bn * FDn * 16 * 4); // 1.5 MB
    float* lbias     = (float*)alloc((size_t)Bn * Hn * 4);
    float* logits    = (float*)alloc((size_t)Bn * Hn * Sn * 4);
    float* probspad  = (float*)alloc((size_t)Bn * 16 * Sn * 4);  // 1 MB
    float* pooledF   = (float*)alloc((size_t)Bn * 16 * FDn * 4); // 1.5 MB
    float* pooledVal = (float*)alloc((size_t)Bn * 16 * En * 4);
    float* ctx       = (float*)alloc((size_t)Bn * En * 4);
    float* attn      = (float*)alloc((size_t)Bn * En * 4);
    float* outval    = (float*)alloc((size_t)Bn * 4);
    (void)ws_size;

    // ---- pre-MLP: scores over all B*S rows (big WMMA GEMMs) ----
    gemm_wmma_bf16<8><<<dim3(2, 128), 256, 0, stream>>>(vf, VDn, pre_w1, 256, pre_b1,
                                                        h1, 256, MROWS, VDn, 256, 1);
    gemm_wmma_bf16<2><<<dim3(1, 128), 256, 0, stream>>>(h1, 256, pre_w2, 32, pre_b2,
                                                        h2, 32, MROWS, 256, 32, 1);
    score3_kernel<<<MROWS / 256, 256, 0, stream>>>(h2, pre_w3, pre_b3, scores, MROWS);

    // ---- top-k select + gather ----
    topk_kernel<<<Bn, Sn, 0, stream>>>(scores, topk_idx);
    gather_kernel<<<dim3(KPAD, Bn), 256, 0, stream>>>(vf, topk_idx, topk_v);

    // ---- CAM ----
    gemm_wmma_bf16<8><<<dim3(4, 14), 256, 0, stream>>>(topk_v, VDn, ft_w, CAMn, ft_b,
                                                       t_buf, CAMn, MCAM, VDn, CAMn, 1);
    gemm_wmma_bf16<8><<<dim3(4, 14), 256, 0, stream>>>(t_buf, CAMn, fp_w, CAMn, fp_b,
                                                       c_buf, CAMn, MCAM, CAMn, CAMn, 1);
    cam_agg_kernel<<<Bn, 256, 0, stream>>>(c_buf, agg_w, agg_b, cam);

    // ---- query path (tiny GEMMs) ----
    gemm_wmma_bf16<8><<<dim3(4, 1), 256, 0, stream>>>(cam, CAMn, q_w, En, q_b,
                                                      q1, En, Bn, CAMn, En, 0);
    gemm_wmma_bf16<8><<<dim3(4, 1), 256, 0, stream>>>(q1, En, in_w, 3 * En, in_b,
                                                      qq, En, Bn, En, En, 0);
    qhead_w1_kernel<<<Bn * Hn, 512, 0, stream>>>(in_w, qq, w1v);
    ukernel<<<Bn * Hn, 256, 0, stream>>>(k_w, k_b, in_b, qq, w1v, u2pad, lbias);

    // ---- attention (q_len==1): logits -> softmax -> pooled ----
    logits_wmma_kernel<<<dim3(Sn / 128, Bn), 256, 0, stream>>>(vf, tf, u2pad, lbias, logits);
    softmax_pad_kernel<<<Bn * 16, 256, 0, stream>>>(logits, probspad);
    pooled_wmma_kernel<<<dim3(FDn / 128, Bn), 256, 0, stream>>>(probspad, vf, tf, pooledF);

    // ---- value path: pooledVal = pooledF @ v_w + v_b  (M = 256 rows) ----
    gemm_wmma_bf16<8><<<dim3(4, 2), 256, 0, stream>>>(pooledF, FDn, v_w, En, v_b,
                                                      pooledVal, En, Bn * 16, FDn, En, 0);
    ctx_kernel<<<Bn * Hn, 128, 0, stream>>>(pooledVal, in_w, in_b, ctx);

    // ---- output projection + post-MLP (once per batch) + broadcast ----
    gemm_wmma_bf16<8><<<dim3(4, 1), 256, 0, stream>>>(ctx, En, out_w, En, out_b,
                                                      attn, En, Bn, En, En, 0);
    postmlp_kernel<<<Bn, 256, 0, stream>>>(attn, post_w1, post_b1, post_w2, post_b2,
                                           post_w3, post_b3, outval);
    bcast_kernel<<<MROWS / 256, 256, 0, stream>>>(outval, (float*)d_out, MROWS);
}
